// VAE_45698452029583
// MI455X (gfx1250) — matrix operand order repaired and checked by reading, not compiled, not run on hardware
//
#include <hip/hip_runtime.h>
#include <hip/hip_bf16.h>

// ---------------------------------------------------------------------------
// Graph-VAE forward for MI455X (gfx1250, wave32).
//   - Dense GEMMs via v_wmma_f32_16x16x32_f16 (f16 in, f32 acc), branchless
//     vectorized LDS staging (scalar f32 global loads -> v8h ds_store_b128)
//   - Edge segment-sums via per-edge float atomics (atomic-bound, ~8GB traffic)
//   - Exact top-k via u64-key bitonic sort (desc score, asc index tie-break)
// ---------------------------------------------------------------------------

#define N_N 100000
#define N_E 1600000
#define N_H 128
#define K1_ 50000
#define K2_ 25000
#define K3_ 12500

#define CDIV(a, b) (((a) + (b) - 1) / (b))

typedef __attribute__((ext_vector_type(16))) _Float16 v16h;
typedef __attribute__((ext_vector_type(8)))  _Float16 v8h;
typedef __attribute__((ext_vector_type(8)))  float    v8f;
typedef __attribute__((ext_vector_type(4)))  unsigned v4u;

__device__ __forceinline__ float lrelu(float v) { return v > 0.f ? v : 0.2f * v; }

__device__ __forceinline__ void atomicMaxF(float* addr, float v) {
    int* ai = (int*)addr;
    int old = __float_as_int(*addr);
    while (__int_as_float(old) < v) {
        int prev = atomicCAS(ai, old, __float_as_int(v));
        if (prev == old) break;
        old = prev;
    }
}

// ---------------------------------------------------------------------------
// WMMA GEMM: C[M x Nc] = A[M x K] @ B[K x Nc], f32 global, f16 compute.
// Block = 128 threads = 4 waves; block tile 32x32(out) x 64(K); two wmma per
// K-step. A overread past M is safe (A always lives in the big ws buffers);
// rows >= M are masked only at the C store.
// Requires: K % 64 == 0, Nc % 32 == 0.
// ---------------------------------------------------------------------------
__global__ __launch_bounds__(128) void gemm_wmma_f16(
    const float* __restrict__ A, const float* __restrict__ B,
    float* __restrict__ C, int M, int K, int Nc) {
    __shared__ _Float16 As[32 * 64];   // [m][k]
    __shared__ _Float16 Bs[32 * 64];   // [n][k]  (transposed: K contiguous)
    const int t = threadIdx.x;
    const int mBase = blockIdx.y * 32;
    const int nBase = blockIdx.x * 32;
    const int lane = t & 31;
    const int wave = t >> 5;
    const int mr = (wave >> 1) << 4;   // wave's 16-row sub-tile
    const int nc = (wave & 1) << 4;    // wave's 16-col sub-tile
    const int g  = lane >> 4;          // lane half-group
    const int ml = lane & 15;
    // staging assignment (fixed per thread, branchless)
    const int ar = t >> 2;             // A row 0..31
    const int ac = (t & 3) << 4;       // A col group {0,16,32,48}
    const int bn = t & 31;             // B column (n) 0..31
    const int bk = (t >> 5) << 4;      // B k group {0,16,32,48}

    v8f acc = {};
    for (int kb = 0; kb < K; kb += 64) {
        // ---- load A chunk: 16 contiguous floats of row (mBase+ar) ----
        const float* ap = A + (size_t)(mBase + ar) * K + kb + ac;
        v8h pa0, pa1;
#pragma unroll
        for (int j = 0; j < 8; ++j) pa0[j] = (_Float16)ap[j];
#pragma unroll
        for (int j = 0; j < 8; ++j) pa1[j] = (_Float16)ap[8 + j];
        // ---- load B chunk: 16 k-strided floats of column (nBase+bn) ----
        const float* bp = B + (size_t)(kb + bk) * Nc + nBase + bn;
        v8h pb0, pb1;
#pragma unroll
        for (int j = 0; j < 8; ++j) pb0[j] = (_Float16)bp[(size_t)j * Nc];
#pragma unroll
        for (int j = 0; j < 8; ++j) pb1[j] = (_Float16)bp[(size_t)(8 + j) * Nc];

        __syncthreads();   // previous iteration's fragment reads are done
        *(v8h*)(As + ar * 64 + ac)      = pa0;
        *(v8h*)(As + ar * 64 + ac + 8)  = pa1;
        *(v8h*)(Bs + bn * 64 + bk)      = pb0;
        *(v8h*)(Bs + bn * 64 + bk + 8)  = pb1;
        __syncthreads();

        const v4u* Ap = (const v4u*)(As + (mr + ml) * 64);
        const v4u* Bp = (const v4u*)(Bs + (nc + ml) * 64);
#pragma unroll
        for (int kk = 0; kk < 2; ++kk) {
            union { v4u q[2]; v16h v; } fa, fb;
            // A fragment: halves kk*32 + {8g..8g+7, 16+8g..16+8g+7}
            fa.q[0] = Ap[kk * 4 + g];
            fa.q[1] = Ap[kk * 4 + 2 + g];
            // B fragment: halves kk*32 + {8g..8g+7, 16+8g..16+8g+7}
            fb.q[0] = Bp[kk * 4 + g];
            fb.q[1] = Bp[kk * 4 + 2 + g];
            acc = __builtin_amdgcn_wmma_f32_16x16x32_f16(
                false, fa.v, false, fb.v, (short)0, acc, false, false);
        }
    }
#pragma unroll
    for (int r = 0; r < 8; ++r) {
        int row = mBase + mr + r + 8 * g;   // C layout: VGPR r, lane group g
        if (row < M) C[(size_t)row * Nc + nBase + nc + ml] = acc[r];
    }
}

// ---------------------------------------------------------------------------
// Small kernels
// ---------------------------------------------------------------------------
__global__ void fill_f32(float* p, float v, int n) {
    int i = blockIdx.x * blockDim.x + threadIdx.x;
    if (i < n) p[i] = v;
}
__global__ void fill_i32(int* p, int v, int n) {
    int i = blockIdx.x * blockDim.x + threadIdx.x;
    if (i < n) p[i] = v;
}

// h = x @ W0  (K = 3)
__global__ void gemm_din3(const float* __restrict__ x, const float* __restrict__ W,
                          float* __restrict__ out, int n) {
    int t = blockIdx.x * blockDim.x + threadIdx.x;
    if (t >= n * N_H) return;
    int i = t >> 7, f = t & 127;
    out[t] = x[i * 3] * W[f] + x[i * 3 + 1] * W[N_H + f] + x[i * 3 + 2] * W[2 * N_H + f];
}

// ---- GCN ----
__global__ void deg_edges(const int* __restrict__ dst, const int* __restrict__ msk,
                          float* deg, int E_) {
    int e = blockIdx.x * blockDim.x + threadIdx.x;
    if (e >= E_) return;
    if (msk && !msk[e]) return;
    atomicAdd(&deg[dst[e]], 1.f);
}
__global__ void rsqrt_inplace(float* p, int n) {
    int i = blockIdx.x * blockDim.x + threadIdx.x;
    if (i < n) p[i] = rsqrtf(fmaxf(p[i], 1e-12f));
}
__global__ void gcn_init(const float* __restrict__ h, const float* __restrict__ dis,
                         const float* __restrict__ bias, float* __restrict__ out,
                         int n, int F) {
    int t = blockIdx.x * blockDim.x + threadIdx.x;
    if (t >= n * F) return;
    int i = t / F, f = t % F;
    float d = dis[i];
    out[t] = bias[f] + d * d * h[t];
}
__global__ void gcn_edges(const int* __restrict__ src, const int* __restrict__ dst,
                          const int* __restrict__ msk, const float* __restrict__ h,
                          const float* __restrict__ dis, float* __restrict__ out,
                          int E_, int F) {
    unsigned t = blockIdx.x * blockDim.x + threadIdx.x;
    int chunks = F >> 2;
    unsigned e = t / chunks;
    if (e >= (unsigned)E_) return;
    if (msk && !msk[e]) return;
    int s = src[e], d = dst[e];
    float w = dis[s] * dis[d];
    int c = (t % chunks) << 2;
    const float4 hv = *(const float4*)(h + (size_t)s * F + c);
    float* o = out + (size_t)d * F + c;
    atomicAdd(o + 0, w * hv.x); atomicAdd(o + 1, w * hv.y);
    atomicAdd(o + 2, w * hv.z); atomicAdd(o + 3, w * hv.w);
}

// ---- GAT ----
__global__ void dot2(const float* __restrict__ h, const float* __restrict__ av,
                     const float* __restrict__ bv, float* oa, float* ob, int n, int F) {
    int i = blockIdx.x * blockDim.x + threadIdx.x;
    if (i >= n) return;
    float sa = 0.f, sb = 0.f;
    for (int f = 0; f < F; ++f) {
        float hv = h[(size_t)i * F + f];
        sa += hv * av[f];
        sb += hv * bv[f];
    }
    oa[i] = sa; ob[i] = sb;
}
__global__ void gat_emax_init(const float* oa, const float* ob, float* emax, int n) {
    int i = blockIdx.x * blockDim.x + threadIdx.x;
    if (i < n) emax[i] = lrelu(oa[i] + ob[i]);   // self loop always present
}
__global__ void gat_emax_edges(const int* src, const int* dst, const int* msk,
                               const float* oa, const float* ob, float* emax, int E_) {
    int e = blockIdx.x * blockDim.x + threadIdx.x;
    if (e >= E_) return;
    if (!msk[e]) return;
    atomicMaxF(&emax[dst[e]], lrelu(oa[src[e]] + ob[dst[e]]));
}
__global__ void gat_den_init(const float* oa, const float* ob, const float* emax,
                             float* den, int n) {
    int i = blockIdx.x * blockDim.x + threadIdx.x;
    if (i < n) den[i] = expf(lrelu(oa[i] + ob[i]) - emax[i]);
}
__global__ void gat_den_edges(const int* src, const int* dst, const int* msk,
                              const float* oa, const float* ob, const float* emax,
                              float* den, int E_) {
    int e = blockIdx.x * blockDim.x + threadIdx.x;
    if (e >= E_) return;
    if (!msk[e]) return;
    int s = src[e], d = dst[e];
    atomicAdd(&den[d], expf(lrelu(oa[s] + ob[d]) - emax[d]));
}
__global__ void gat_out_init(const float* __restrict__ h, const float* oa, const float* ob,
                             const float* emax, const float* den, const float* bias,
                             float* __restrict__ out, int n, int F) {
    int t = blockIdx.x * blockDim.x + threadIdx.x;
    if (t >= n * F) return;
    int i = t / F, f = t % F;
    float alpha = expf(lrelu(oa[i] + ob[i]) - emax[i]) / (den[i] + 1e-16f);
    out[t] = bias[f] + alpha * h[t];
}
__global__ void gat_out_edges(const int* __restrict__ src, const int* __restrict__ dst,
                              const int* __restrict__ msk, const float* __restrict__ h,
                              const float* oa, const float* ob, const float* emax,
                              const float* den, float* __restrict__ out, int E_, int F) {
    unsigned t = blockIdx.x * blockDim.x + threadIdx.x;
    int chunks = F >> 2;
    unsigned e = t / chunks;
    if (e >= (unsigned)E_) return;
    if (!msk[e]) return;
    int s = src[e], d = dst[e];
    float alpha = expf(lrelu(oa[s] + ob[d]) - emax[d]) / (den[d] + 1e-16f);
    int c = (t % chunks) << 2;
    const float4 hv = *(const float4*)(h + (size_t)s * F + c);
    float* o = out + (size_t)d * F + c;
    atomicAdd(o + 0, alpha * hv.x); atomicAdd(o + 1, alpha * hv.y);
    atomicAdd(o + 2, alpha * hv.z); atomicAdd(o + 3, alpha * hv.w);
}

// ---- TopK pooling ----
__global__ void score_kernel(const float* __restrict__ h, const float* __restrict__ w,
                             float* score, int n, int F) {
    int i = blockIdx.x * blockDim.x + threadIdx.x;
    if (i >= n) return;
    float nn = 0.f, dp = 0.f;
    for (int f = 0; f < F; ++f) {
        float wv = w[f];
        nn += wv * wv;
        dp += h[(size_t)i * F + f] * wv;
    }
    score[i] = tanhf(dp / (sqrtf(nn) + 1e-16f));
}
__global__ void build_keys(const float* score, unsigned long long* keys, int n, int npow2) {
    int i = blockIdx.x * blockDim.x + threadIdx.x;
    if (i >= npow2) return;
    if (i < n) {
        unsigned u = __float_as_uint(score[i]);
        u = (u >> 31) ? ~u : (u | 0x80000000u);   // monotonic ascending map
        unsigned inv = ~u;                        // descending score
        keys[i] = ((unsigned long long)inv << 32) | (unsigned)i;  // asc idx tie-break
    } else {
        keys[i] = ~0ULL;                          // padding sorts last
    }
}
__global__ void bitonic_step(unsigned long long* keys, int j, int k) {
    int i = blockIdx.x * blockDim.x + threadIdx.x;
    int ixj = i ^ j;
    if (ixj > i) {
        unsigned long long a = keys[i], b = keys[ixj];
        bool asc = ((i & k) == 0);
        if ((a > b) == asc) { keys[i] = b; keys[ixj] = a; }
    }
}
__global__ void extract_topk(const unsigned long long* keys, const float* score,
                             int* perm, float* vals, int k) {
    int i = blockIdx.x * blockDim.x + threadIdx.x;
    if (i >= k) return;
    int idx = (int)(keys[i] & 0xFFFFFFFFu);
    perm[i] = idx;
    vals[i] = score[idx];
}
__global__ void gather_scale(const float* __restrict__ h, const int* perm,
                             const float* vals, float* __restrict__ out, int k, int F) {
    int t = blockIdx.x * blockDim.x + threadIdx.x;
    if (t >= k * F) return;
    int i = t / F, f = t % F;
    out[t] = h[(size_t)perm[i] * F + f] * vals[i];
}
__global__ void set_mapping(const int* perm, int* mapping, int k) {
    int i = blockIdx.x * blockDim.x + threadIdx.x;
    if (i < k) mapping[perm[i]] = i;
}
__global__ void remap_edges(const int* __restrict__ src, const int* __restrict__ dst,
                            const int* __restrict__ mskIn, const int* __restrict__ mapping,
                            int* so, int* dto, int* mo, int E_) {
    int e = blockIdx.x * blockDim.x + threadIdx.x;
    if (e >= E_) return;
    int ok = mskIn ? mskIn[e] : 1;
    int ns = mapping[src[e]], nd = mapping[dst[e]];
    ok = ok && (ns >= 0) && (nd >= 0);
    so[e] = ok ? ns : 0;
    dto[e] = ok ? nd : 0;
    mo[e] = ok;
}

// ---- VAE head: mu/logvar -> z -> latent decode to 128 ----
__global__ void vae_head(const float* __restrict__ h, const float* Wmu, const float* bmu,
                         const float* Wlv, const float* blv, const float* Wld,
                         const float* bld, const float* eps,
                         float* mu_out, float* lv_out, float* __restrict__ zlat, int n) {
    int i = blockIdx.x * blockDim.x + threadIdx.x;
    if (i >= n) return;
    float mu0 = bmu[0], mu1 = bmu[1], lv0 = blv[0], lv1 = blv[1];
    for (int f = 0; f < N_H; ++f) {
        float hv = h[(size_t)i * N_H + f];
        mu0 += hv * Wmu[f * 2 + 0]; mu1 += hv * Wmu[f * 2 + 1];
        lv0 += hv * Wlv[f * 2 + 0]; lv1 += hv * Wlv[f * 2 + 1];
    }
    float z0 = mu0 + eps[i * 2 + 0] * expf(0.5f * lv0);
    float z1 = mu1 + eps[i * 2 + 1] * expf(0.5f * lv1);
    mu_out[i * 2 + 0] = mu0; mu_out[i * 2 + 1] = mu1;
    lv_out[i * 2 + 0] = lv0; lv_out[i * 2 + 1] = lv1;
    for (int f = 0; f < N_H; ++f)
        zlat[(size_t)i * N_H + f] = z0 * Wld[f] + z1 * Wld[N_H + f] + bld[f];
}

// ---------------------------------------------------------------------------
extern "C" void kernel_launch(void* const* d_in, const int* in_sizes, int n_in,
                              void* d_out, int out_size, void* d_ws, size_t ws_size,
                              hipStream_t stream) {
    (void)in_sizes; (void)n_in; (void)out_size; (void)ws_size;
    const float* x   = (const float*)d_in[0];
    const int*   ei  = (const int*)d_in[1];
    const int*   uei = (const int*)d_in[2];
    const float* W0  = (const float*)d_in[4];
    const float* b0  = (const float*)d_in[5];
    const float* W1  = (const float*)d_in[6];
    const float* a1s = (const float*)d_in[7];
    const float* a1d = (const float*)d_in[8];
    const float* b1  = (const float*)d_in[9];
    const float* W2  = (const float*)d_in[10];
    const float* a2s = (const float*)d_in[11];
    const float* a2d = (const float*)d_in[12];
    const float* b2  = (const float*)d_in[13];
    const float* pw1 = (const float*)d_in[14];
    const float* pw2 = (const float*)d_in[15];
    const float* pw3 = (const float*)d_in[16];
    const float* Wmu = (const float*)d_in[17];
    const float* bmu = (const float*)d_in[18];
    const float* Wlv = (const float*)d_in[19];
    const float* blv = (const float*)d_in[20];
    const float* Wld = (const float*)d_in[21];
    const float* bld = (const float*)d_in[22];
    const float* Wd0 = (const float*)d_in[23];
    const float* bd0 = (const float*)d_in[24];
    const float* Wd1 = (const float*)d_in[25];
    const float* bd1 = (const float*)d_in[26];
    const float* Wd2 = (const float*)d_in[27];
    const float* bd2 = (const float*)d_in[28];
    const float* eps = (const float*)d_in[29];
    float* out = (float*)d_out;
    char*  ws  = (char*)d_ws;

    // Workspace layout (~145 MB)
    size_t off = 0;
    float* hA = (float*)(ws + off); off += (size_t)N_N * N_H * 4;
    float* hB = (float*)(ws + off); off += (size_t)N_N * N_H * 4;
    float* fa = (float*)(ws + off); off += (size_t)N_N * 4;   // deg/dis or as_
    float* fb = (float*)(ws + off); off += (size_t)N_N * 4;   // ad_
    float* fc = (float*)(ws + off); off += (size_t)N_N * 4;   // emax
    float* fd = (float*)(ws + off); off += (size_t)N_N * 4;   // den
    float* score = (float*)(ws + off); off += (size_t)N_N * 4;
    float* vals  = (float*)(ws + off); off += (size_t)K1_ * 4;
    int*   perm  = (int*)(ws + off);   off += (size_t)K1_ * 4;
    int*   mapping = (int*)(ws + off); off += (size_t)N_N * 4;
    unsigned long long* keys = (unsigned long long*)(ws + off); off += (size_t)131072 * 8;
    int* eSrcA = (int*)(ws + off); off += (size_t)N_E * 4;
    int* eDstA = (int*)(ws + off); off += (size_t)N_E * 4;
    int* eMskA = (int*)(ws + off); off += (size_t)N_E * 4;
    int* eSrcB = (int*)(ws + off); off += (size_t)N_E * 4;
    int* eDstB = (int*)(ws + off); off += (size_t)N_E * 4;
    int* eMskB = (int*)(ws + off); off += (size_t)N_E * 4;

    auto run_pool = [&](const float* hin, const float* pw, int n, int npow2, int k,
                        const int* srcIn, const int* dstIn, const int* mskIn,
                        float* hout, int* srcOut, int* dstOut, int* mskOut) {
        score_kernel<<<CDIV(n, 256), 256, 0, stream>>>(hin, pw, score, n, N_H);
        build_keys<<<CDIV(npow2, 256), 256, 0, stream>>>(score, keys, n, npow2);
        for (int kk = 2; kk <= npow2; kk <<= 1)
            for (int j = kk >> 1; j > 0; j >>= 1)
                bitonic_step<<<npow2 / 256, 256, 0, stream>>>(keys, j, kk);
        extract_topk<<<CDIV(k, 256), 256, 0, stream>>>(keys, score, perm, vals, k);
        gather_scale<<<CDIV(k * N_H, 256), 256, 0, stream>>>(hin, perm, vals, hout, k, N_H);
        fill_i32<<<CDIV(n, 256), 256, 0, stream>>>(mapping, -1, n);
        set_mapping<<<CDIV(k, 256), 256, 0, stream>>>(perm, mapping, k);
        remap_edges<<<CDIV(N_E, 256), 256, 0, stream>>>(srcIn, dstIn, mskIn, mapping,
                                                        srcOut, dstOut, mskOut, N_E);
    };

    auto run_gcn_agg = [&](const float* hin, const float* dis, const float* bias, float* o,
                           const int* src, const int* dst, const int* msk, int n, int F) {
        gcn_init<<<CDIV(n * F, 256), 256, 0, stream>>>(hin, dis, bias, o, n, F);
        gcn_edges<<<CDIV(N_E * (F >> 2), 256), 256, 0, stream>>>(src, dst, msk, hin, dis,
                                                                 o, N_E, F);
    };

    auto run_gat = [&](const float* hlin, const float* asv, const float* adv,
                       const float* bias, float* o, const int* src, const int* dst,
                       const int* msk, int n) {
        dot2<<<CDIV(n, 256), 256, 0, stream>>>(hlin, asv, adv, fa, fb, n, N_H);
        gat_emax_init<<<CDIV(n, 256), 256, 0, stream>>>(fa, fb, fc, n);
        gat_emax_edges<<<CDIV(N_E, 256), 256, 0, stream>>>(src, dst, msk, fa, fb, fc, N_E);
        gat_den_init<<<CDIV(n, 256), 256, 0, stream>>>(fa, fb, fc, fd, n);
        gat_den_edges<<<CDIV(N_E, 256), 256, 0, stream>>>(src, dst, msk, fa, fb, fc, fd, N_E);
        gat_out_init<<<CDIV(n * N_H, 256), 256, 0, stream>>>(hlin, fa, fb, fc, fd, bias, o,
                                                             n, N_H);
        gat_out_edges<<<CDIV(N_E * 32, 256), 256, 0, stream>>>(src, dst, msk, hlin, fa, fb,
                                                               fc, fd, o, N_E, N_H);
    };

    // 1) hA = x @ W0 (K=3, plain kernel)
    gemm_din3<<<CDIV(N_N * N_H, 256), 256, 0, stream>>>(x, W0, hA, N_N);

    // 2) GCN level 0 on directed edges (mask = all true)
    fill_f32<<<CDIV(N_N, 256), 256, 0, stream>>>(fa, 1.f, N_N);          // self-loop deg
    deg_edges<<<CDIV(N_E, 256), 256, 0, stream>>>(ei + N_E, nullptr, fa, N_E);
    rsqrt_inplace<<<CDIV(N_N, 256), 256, 0, stream>>>(fa, N_N);
    run_gcn_agg(hA, fa, b0, hB, ei, ei + N_E, nullptr, N_N, N_H);        // h0 in hB

    // 3) Pool 1 (undirected edges, mask all-true) -> hA, edges A
    run_pool(hB, pw1, N_N, 131072, K1_, uei, uei + N_E, nullptr,
             hA, eSrcA, eDstA, eMskA);

    // 4) GAT 1: hB = hA @ W1 (WMMA), aggregate -> hA
    gemm_wmma_f16<<<dim3(N_H / 32, CDIV(K1_, 32)), 128, 0, stream>>>(hA, W1, hB,
                                                                     K1_, N_H, N_H);
    run_gat(hB, a1s, a1d, b1, hA, eSrcA, eDstA, eMskA, K1_);

    // 5) Pool 2 -> hB, edges B
    run_pool(hA, pw2, K1_, 65536, K2_, eSrcA, eDstA, eMskA,
             hB, eSrcB, eDstB, eMskB);

    // 6) GAT 2: hA = hB @ W2 (WMMA), aggregate -> hB
    gemm_wmma_f16<<<dim3(N_H / 32, CDIV(K2_, 32)), 128, 0, stream>>>(hB, W2, hA,
                                                                     K2_, N_H, N_H);
    run_gat(hA, a2s, a2d, b2, hB, eSrcB, eDstB, eMskB, K2_);

    // 7) Pool 3 -> hA, edges A
    run_pool(hB, pw3, K2_, 32768, K3_, eSrcB, eDstB, eMskB,
             hA, eSrcA, eDstA, eMskA);

    // 8) VAE head: mu/logvar straight into d_out slices; zlat -> hB (K3 x 128)
    float* mu_out = out + (size_t)K3_ * 32;
    float* lv_out = out + (size_t)K3_ * 32 + (size_t)K3_ * 2;
    vae_head<<<CDIV(K3_, 256), 256, 0, stream>>>(hA, Wmu, bmu, Wlv, blv, Wld, bld, eps,
                                                 mu_out, lv_out, hB, K3_);

    // 9) Decoder graph normalization (same for all 3 decoder GCNs)
    fill_f32<<<CDIV(K3_, 256), 256, 0, stream>>>(fa, 1.f, K3_);
    deg_edges<<<CDIV(N_E, 256), 256, 0, stream>>>(eDstA, eMskA, fa, N_E);
    rsqrt_inplace<<<CDIV(K3_, 256), 256, 0, stream>>>(fa, K3_);

    // 10) GCN d0: 128 -> 128
    gemm_wmma_f16<<<dim3(128 / 32, CDIV(K3_, 32)), 128, 0, stream>>>(hB, Wd0, hA,
                                                                     K3_, 128, 128);
    run_gcn_agg(hA, fa, bd0, hB, eSrcA, eDstA, eMskA, K3_, 128);

    // 11) GCN d1: 128 -> 64
    gemm_wmma_f16<<<dim3(64 / 32, CDIV(K3_, 32)), 128, 0, stream>>>(hB, Wd1, hA,
                                                                    K3_, 128, 64);
    run_gcn_agg(hA, fa, bd1, hB, eSrcA, eDstA, eMskA, K3_, 64);

    // 12) GCN d2: 64 -> 32, writes z directly into d_out
    gemm_wmma_f16<<<dim3(32 / 32, CDIV(K3_, 32)), 128, 0, stream>>>(hB, Wd2, hA,
                                                                    K3_, 64, 32);
    run_gcn_agg(hA, fa, bd2, out, eSrcA, eDstA, eMskA, K3_, 32);
}
